// MultiHeadAttention_23545010717014
// MI455X (gfx1250) — matrix operand order repaired and checked by reading, not compiled, not run on hardware
//
#include <hip/hip_runtime.h>
#include <hip/hip_bf16.h>

typedef __bf16 bf16_t;
typedef __attribute__((ext_vector_type(16))) __bf16 v16bf;
typedef __attribute__((ext_vector_type(8)))  __bf16 v8bf;
typedef __attribute__((ext_vector_type(8)))  float  v8f;

#define DMODEL 512
#define NHEAD  8
#define DKDIM  64
#define SLEN   1024
#define BATCH  8

__device__ __forceinline__ bf16_t f2bf(float x) {
  union { float f; unsigned u; } a; a.f = x;
  unsigned r = a.u + 0x7FFFu + ((a.u >> 16) & 1u);   // round-to-nearest-even
  union { unsigned short s; bf16_t b; } c; c.s = (unsigned short)(r >> 16);
  return c.b;
}

__device__ __forceinline__ v8f vzero8() {
  v8f z;
#pragma unroll
  for (int i = 0; i < 8; ++i) z[i] = 0.0f;
  return z;
}

__device__ __forceinline__ v16bf cat8(v8bf lo, v8bf hi) {
  return __builtin_shufflevector(lo, hi, 0,1,2,3,4,5,6,7,8,9,10,11,12,13,14,15);
}

__global__ void convert_f32_bf16(const float* __restrict__ src, bf16_t* __restrict__ dst, int n) {
  int i = blockIdx.x * blockDim.x + threadIdx.x;
  int stride = gridDim.x * blockDim.x;
  for (; i < n; i += stride) dst[i] = f2bf(src[i]);
}

// C[M,N] = A[M,K] * W[N,K]^T + bias[N]
// MODE 0: bf16 out, scatter to [b,h,s,d]   (Q, K)
// MODE 1: bf16 out, scatter to [b,h,d,s]   (V transposed)
// MODE 2: f32 out, row-major [M,N]         (final projection)
// wave computes 16(M) x 64(N); block = 8 waves -> 128 x 64 tile
template <int MODE>
__global__ void gemm_wmma(const bf16_t* __restrict__ A, const bf16_t* __restrict__ W,
                          const float* __restrict__ bias, void* __restrict__ out,
                          int M, int N, int K) {
  const int wave = threadIdx.x >> 5;
  const int lane = threadIdx.x & 31;
  const int lm = lane & 15;
  const int hi = lane >> 4;
  const int m0 = (blockIdx.x * 8 + wave) * 16;
  const int n0 = blockIdx.y * 64;

  v8f acc[4];
#pragma unroll
  for (int s = 0; s < 4; ++s) acc[s] = vzero8();

  const bf16_t* pA = A + (size_t)(m0 + lm) * K;
  for (int kk = 0; kk < K; kk += 32) {
    // A fragment 16x32 (M x K), contiguous k = hi*16 + e
    v16bf a = *(const v16bf*)(pA + kk + hi * 16);
#pragma unroll
    for (int s = 0; s < 4; ++s) {
      // B fragment 32x16 (K x N): B[k][n] = W[n][k]; lane n = lm, k = hi*16 + e
      const bf16_t* pW = W + (size_t)(n0 + s * 16 + lm) * K + kk + hi * 16;
      v16bf b = *(const v16bf*)pW;
      acc[s] = __builtin_amdgcn_wmma_f32_16x16x32_bf16(false, a, false, b,
                                                       (short)0, acc[s], false, false);
    }
  }

#pragma unroll
  for (int s = 0; s < 4; ++s) {
    int col = n0 + s * 16 + lm;
    float bv = bias[col];
#pragma unroll
    for (int v = 0; v < 8; ++v) {
      int m = m0 + hi * 8 + v;              // C/D layout: row = v + hi*8
      float val = acc[s][v] + bv;
      if (MODE == 2) {
        ((float*)out)[(size_t)m * N + col] = val;
      } else {
        int bb = m >> 10, ss = m & 1023;
        int hh = col >> 6, dd = col & 63;
        size_t idx;
        if (MODE == 0)
          idx = (((size_t)(bb * NHEAD + hh)) * SLEN + ss) * DKDIM + dd;
        else
          idx = (((size_t)(bb * NHEAD + hh)) * DKDIM + dd) * SLEN + ss;
        ((bf16_t*)out)[idx] = f2bf(val);
      }
    }
  }
}

// Flash attention, one wave per 16-query tile.
// Computes S^T = K_tile * Q^T  (so q lives in the lane index of the C fragment),
// then O^T = V^T * P^T. Q,K: [b,h,s,64] bf16; Vt: [b,h,64,s] bf16.
// ctx out: [b, s, h*64+d] bf16.
__global__ void attention_wmma(const bf16_t* __restrict__ Q, const bf16_t* __restrict__ K,
                               const bf16_t* __restrict__ Vt, const float* __restrict__ mask,
                               bf16_t* __restrict__ ctx) {
  const int lane = threadIdx.x & 31;
  const int lm = lane & 15;
  const int hi = lane >> 4;
  const int wave = threadIdx.x >> 5;
  const int b = blockIdx.z, h = blockIdx.y;
  const int q0 = (blockIdx.x * 4 + wave) * 16;

  const bf16_t* Qh = Q  + ((size_t)(b * NHEAD + h)) * SLEN * DKDIM;
  const bf16_t* Kh = K  + ((size_t)(b * NHEAD + h)) * SLEN * DKDIM;
  const bf16_t* Vh = Vt + ((size_t)(b * NHEAD + h)) * DKDIM * SLEN;
  const float*  mrow = mask + ((size_t)b * SLEN + (q0 + lm)) * SLEN;  // this lane's query row

  // Loop-invariant B fragments of Q^T (k = d, n = q): lane n = lm, d = dhalf*32 + hi*16 + e
  v16bf bq0 = *(const v16bf*)(Qh + (size_t)(q0 + lm) * DKDIM + hi * 16);
  v16bf bq1 = *(const v16bf*)(Qh + (size_t)(q0 + lm) * DKDIM + 32 + hi * 16);

  v8f o[4];
#pragma unroll
  for (int t = 0; t < 4; ++t) o[t] = vzero8();
  float Mrow = -3.0e38f, Lrow = 0.0f;

  for (int key0 = 0; key0 < SLEN; key0 += 32) {
    // ---- S^T tiles: keys [key0, key0+16) and [key0+16, key0+32) ----
    v8f st[2];
#pragma unroll
    for (int t = 0; t < 2; ++t) {
      const bf16_t* pK = Kh + (size_t)(key0 + t * 16 + lm) * DKDIM;  // A row = key
      v16bf a0 = *(const v16bf*)(pK + hi * 16);
      v16bf a1 = *(const v16bf*)(pK + 32 + hi * 16);
      v8f s = vzero8();
      s = __builtin_amdgcn_wmma_f32_16x16x32_bf16(false, a0, false, bq0, (short)0, s, false, false);
      s = __builtin_amdgcn_wmma_f32_16x16x32_bf16(false, a1, false, bq1, (short)0, s, false, false);
      st[t] = s;
    }

    // ---- scale + mask; per-query (lane-local) max ----
    float oneM[2][8];
    float tileMax = -3.0e38f;
#pragma unroll
    for (int t = 0; t < 2; ++t) {
#pragma unroll
      for (int v = 0; v < 8; ++v) {
        int key = key0 + t * 16 + hi * 8 + v;      // C/D row = key within tile
        float mv = mrow[key];
        float s = st[t][v] * 0.125f;               // 1/sqrt(64)
        s = s * (1.0f - mv) + mv * (-1.0e7f);
        st[t][v] = s;
        oneM[t][v] = 1.0f - mv;
        tileMax = fmaxf(tileMax, s);
      }
    }
    tileMax = fmaxf(tileMax, __shfl_xor(tileMax, 16, 32));
    float Mnew = fmaxf(Mrow, tileMax);
    float corr = __expf(Mrow - Mnew);

    float p0[8], p1[8];
    float ls = 0.0f;
#pragma unroll
    for (int v = 0; v < 8; ++v) {
      float e0 = __expf(st[0][v] - Mnew);
      float e1 = __expf(st[1][v] - Mnew);
      ls += e0 + e1;                                // denominator keeps masked exps (matches ref)
      p0[v] = e0 * oneM[0][v];                      // numerator zeroed at masked positions
      p1[v] = e1 * oneM[1][v];
    }
    ls += __shfl_xor(ls, 16, 32);
    Lrow = Lrow * corr + ls;
    Mrow = Mnew;

    // ---- build P^T B-fragment (k = key rel 0..31, n = q = lm) ----
    // B half e: k = hi*16 + e. Value (k,q) lives in st[k>>4] VGPR (k&7) at half ((k>>3)&1).
    v16bf bp;
#pragma unroll
    for (int v = 0; v < 8; ++v) {
      float t0 = __shfl_xor(p0[v], 16, 32);
      float t1 = __shfl_xor(p1[v], 16, 32);
      float eLo = hi ? t1 : p0[v];                  // e = v
      float eHi = hi ? p1[v] : t0;                  // e = v + 8
      bp[v]     = f2bf(eLo);
      bp[v + 8] = f2bf(eHi);
    }

    // ---- O^T += V^T * P^T  (4 d-subtiles of 16) ----
#pragma unroll
    for (int t = 0; t < 4; ++t) {
      const bf16_t* pV = Vh + (size_t)(t * 16 + lm) * SLEN + key0;   // A row = d
      v16bf av = *(const v16bf*)(pV + hi * 16);
#pragma unroll
      for (int v = 0; v < 8; ++v) o[t][v] *= corr;                   // corr lane-uniform in q
      o[t] = __builtin_amdgcn_wmma_f32_16x16x32_bf16(false, av, false, bp,
                                                     (short)0, o[t], false, false);
    }
  }

  float inv = 1.0f / Lrow;
#pragma unroll
  for (int t = 0; t < 4; ++t) {
#pragma unroll
    for (int v = 0; v < 8; ++v) {
      int d = t * 16 + hi * 8 + v;                  // O^T row = d, lane col = q
      size_t idx = ((size_t)(b * SLEN + q0 + lm)) * (NHEAD * DKDIM) + h * DKDIM + d;
      ctx[idx] = f2bf(o[t][v] * inv);
    }
  }
}

extern "C" void kernel_launch(void* const* d_in, const int* in_sizes, int n_in,
                              void* d_out, int out_size, void* d_ws, size_t ws_size,
                              hipStream_t stream) {
  (void)in_sizes; (void)n_in; (void)out_size; (void)ws_size;
  const float* x    = (const float*)d_in[0];
  const float* mask = (const float*)d_in[1];
  const float* wq   = (const float*)d_in[2];
  const float* bq   = (const float*)d_in[3];
  const float* wk   = (const float*)d_in[4];
  const float* bk   = (const float*)d_in[5];
  const float* wv   = (const float*)d_in[6];
  const float* bv   = (const float*)d_in[7];
  const float* wo   = (const float*)d_in[8];
  const float* bo   = (const float*)d_in[9];
  float* out = (float*)d_out;

  const int M = BATCH * SLEN;  // 8192
  char* ws = (char*)d_ws;
  size_t off = 0;
  auto alloc = [&](size_t bytes) -> void* {
    void* p = ws + off;
    off += (bytes + 255) & ~(size_t)255;
    return p;
  };
  bf16_t* Xb  = (bf16_t*)alloc((size_t)M * DMODEL * 2);
  bf16_t* Wqb = (bf16_t*)alloc((size_t)DMODEL * DMODEL * 2);
  bf16_t* Wkb = (bf16_t*)alloc((size_t)DMODEL * DMODEL * 2);
  bf16_t* Wvb = (bf16_t*)alloc((size_t)DMODEL * DMODEL * 2);
  bf16_t* Wob = (bf16_t*)alloc((size_t)DMODEL * DMODEL * 2);
  bf16_t* Qb  = (bf16_t*)alloc((size_t)M * DMODEL * 2);
  bf16_t* Kb  = (bf16_t*)alloc((size_t)M * DMODEL * 2);
  bf16_t* Vtb = (bf16_t*)alloc((size_t)M * DMODEL * 2);
  bf16_t* Ctx = (bf16_t*)alloc((size_t)M * DMODEL * 2);

  const int nX = M * DMODEL;
  const int nW = DMODEL * DMODEL;
  convert_f32_bf16<<<(nX + 255) / 256, 256, 0, stream>>>(x,  Xb,  nX);
  convert_f32_bf16<<<(nW + 255) / 256, 256, 0, stream>>>(wq, Wqb, nW);
  convert_f32_bf16<<<(nW + 255) / 256, 256, 0, stream>>>(wk, Wkb, nW);
  convert_f32_bf16<<<(nW + 255) / 256, 256, 0, stream>>>(wv, Wvb, nW);
  convert_f32_bf16<<<(nW + 255) / 256, 256, 0, stream>>>(wo, Wob, nW);

  dim3 gg(M / 128, DMODEL / 64);  // (64, 8) blocks of 256 threads
  gemm_wmma<0><<<gg, 256, 0, stream>>>(Xb, Wqb, bq, Qb,  M, DMODEL, DMODEL);
  gemm_wmma<0><<<gg, 256, 0, stream>>>(Xb, Wkb, bk, Kb,  M, DMODEL, DMODEL);
  gemm_wmma<1><<<gg, 256, 0, stream>>>(Xb, Wvb, bv, Vtb, M, DMODEL, DMODEL);

  dim3 ga(SLEN / 64, NHEAD, BATCH);  // 4 q-tiles per block (4 waves)
  attention_wmma<<<ga, 128, 0, stream>>>(Qb, Kb, Vtb, mask, Ctx);

  gemm_wmma<2><<<gg, 256, 0, stream>>>(Ctx, Wob, bo, out, M, DMODEL, DMODEL);
}
